// AttentionModule_20521353741040
// MI455X (gfx1250) — matrix operand order repaired and checked by reading, not compiled, not run on hardware
//
#include <hip/hip_runtime.h>
#include <hip/hip_bf16.h>

// ---------------------------------------------------------------------------
// CDNA5 (gfx1250, wave32) attention module:
//   rmsnorm -> (QKV GEMMs, WMMA bf16) -> RoPE -> flash attention (WMMA + TDM)
//   -> out GEMM.  All matmuls: v_wmma_f32_16x16x32_bf16 (fp32 accumulate).
//   K tiles in flash attention are staged LDS-side by the Tensor Data Mover
//   (tensor_load_to_lds, TENSORcnt) when the builtin is available.
// ---------------------------------------------------------------------------

typedef __attribute__((ext_vector_type(16))) __bf16 v16bf;
typedef __attribute__((ext_vector_type(8)))  __bf16 v8bf;
typedef __attribute__((ext_vector_type(8)))  float  v8f;
typedef __attribute__((ext_vector_type(4)))  unsigned v4u;
typedef __attribute__((ext_vector_type(8)))  int      v8i;
typedef __attribute__((ext_vector_type(4)))  int      v4i;

#define SEQ_ 4096
#define D_   2048
#define NH_  16
#define HD_  128

#if __has_builtin(__builtin_amdgcn_tensor_load_to_lds) && __has_builtin(__builtin_amdgcn_s_wait_tensorcnt)
#define HAVE_TDM 1
#else
#define HAVE_TDM 0
#endif

static __device__ __forceinline__ __bf16 tobf(float f) { return (__bf16)f; }

static __device__ __forceinline__ unsigned pack2bf(float a, float b) {
    union { __bf16 h[2]; unsigned u; } x;
    x.h[0] = (__bf16)a; x.h[1] = (__bf16)b;
    return x.u;
}

// Low 32 bits of a generic pointer into LDS == LDS byte offset (ISA §10.2
// aperture mapping).
static __device__ __forceinline__ unsigned lds_byte_off(const void* p) {
    return (unsigned)(size_t)p;
}

#if HAVE_TDM
// TDM: async-load a 2-D bf16 tile [rows x 128] (row stride 128 elems) from
// global into LDS, inserting a 16-byte pad after every 256 bytes so the LDS
// row stride is 136 elements (= our sK[32][136] layout).  ISA §8 D# layout.
static __device__ __forceinline__ void tdm_load_k_tile(const __bf16* gsrc,
                                                       unsigned lds_off,
                                                       int rows)
{
    const unsigned long long ga = (unsigned long long)(size_t)gsrc;
    v4u g0;
    g0.x = 1u;                                         // count=1 (user D#)
    g0.y = lds_off;                                    // lds_addr
    g0.z = (unsigned)(ga & 0xFFFFFFFFu);               // global_addr[31:0]
    g0.w = (unsigned)((ga >> 32) & 0x01FFFFFFu)        // global_addr[56:32]
         | (2u << 30);                                 // type=2 ("image")
    v8i g1;
    g1[0] = (int)((1u << 16)        // data_size = 1 -> 2 bytes
                | (1u << 20)        // pad_enable
                | (5u << 22)        // pad_interval: 64 DWORDs (256B)
                | (3u << 25));      // pad_amount:   4 DWORDs  (16B)
    g1[1] = (int)(128u << 16);      // tensor_dim0[15:0] = 128  (bits 79:48)
    g1[2] = (int)((unsigned)SEQ_ << 16); // dim0 hi = 0 | tensor_dim1 lo = SEQ
    g1[3] = (int)(128u << 16);      // tensor_dim1 hi = 0 | tile_dim0 = 128
    g1[4] = rows;                   // tile_dim1 = rows, tile_dim2 = 0
    g1[5] = 128;                    // tensor_dim0_stride (elements)
    g1[6] = 0;
    g1[7] = 0;
    v4i g2 = {};                    // 2-D tensor: groups 2/3 unused (zero)
    v4i g3 = {};
#if __clang_major__ >= 23
    v8i gz = {};
    __builtin_amdgcn_tensor_load_to_lds(g0, g1, g2, g3, gz, 0);
#else
    __builtin_amdgcn_tensor_load_to_lds(g0, g1, g2, g3, 0);
#endif
}
#endif

// ---------------------------------------------------------------------------
// Kernel 1: RMSNorm, fp32 in -> bf16 out (row-major S x D)
// ---------------------------------------------------------------------------
__global__ __launch_bounds__(256) void rmsnorm_kernel(
    const float* __restrict__ xs, const float* __restrict__ w,
    __bf16* __restrict__ h, int D)
{
    const int row = blockIdx.x;
    const float* x = xs + (size_t)row * D;
    float ss = 0.f;
    for (int i = threadIdx.x; i < D; i += 256) { float v = x[i]; ss += v * v; }
    for (int off = 16; off; off >>= 1) ss += __shfl_xor(ss, off, 32);
    __shared__ float red[8];
    __shared__ float rstd;
    if ((threadIdx.x & 31) == 0) red[threadIdx.x >> 5] = ss;
    __syncthreads();
    if (threadIdx.x == 0) {
        float t = 0.f;
        #pragma unroll
        for (int i = 0; i < 8; i++) t += red[i];
        rstd = rsqrtf(t / (float)D + 1e-6f);
    }
    __syncthreads();
    const float r = rstd;
    __bf16* ho = h + (size_t)row * D;
    for (int i = threadIdx.x; i < D; i += 256) ho[i] = tobf(x[i] * r * w[i]);
}

// ---------------------------------------------------------------------------
// Kernel 2: generic WMMA bf16 GEMM.  C(MxN,f32) = A(MxK,bf16) * B(KxN)
//   B fp32 in global, converted to bf16 during LDS staging.
//   BTRANS==0: B row-major KxN.   BTRANS==1: B row-major NxK.
//   SMODE==0 : store fp32 row-major MxN.  SMODE==1: bf16 per-head layout.
// Block: 256 threads = 8 waves arranged 4(M) x 2(N).
// Block tile BM=128, BN=128, BK=32; wave tile 32x64:
//   2 A-frags x 4 B-frags -> 8 WMMAs per K-step from 6 fragment loads.
// ---------------------------------------------------------------------------
template <int BTRANS, int SMODE>
__global__ __launch_bounds__(256) void gemm_bf16_wmma(
    const __bf16* __restrict__ A, const float* __restrict__ B,
    float* __restrict__ Cf, __bf16* __restrict__ Cbh,
    int M, int N, int K)
{
    constexpr int BM = 128, BN = 128, BK = 32, PAD = 16; // 96B row stride
    __shared__ __bf16 sA[BM][BK + PAD];
    __shared__ __bf16 sBt[BN][BK + PAD];   // B tile transposed: [n][k]

    const int m0 = blockIdx.y * BM;
    const int n0 = blockIdx.x * BN;
    const int tid  = threadIdx.x;
    const int lane = tid & 31;
    const int wave = tid >> 5;
    const int waveM = wave & 3;            // 0..3 -> 32-row strip
    const int waveN = wave >> 2;           // 0..1 -> 64-col strip
    const int hi16 = lane >> 4;
    const int l16  = lane & 15;

    // per-thread staging coordinates (A: 128x32 tile, 16 elems/thread)
    const int arow = tid >> 1;
    const int acb  = (tid & 1) * 16;

    v8f acc[2][4] = {};

    for (int k0 = 0; k0 < K; k0 += BK) {
        // ---- stage A: 128x32 bf16, 32B per thread, fully vectorized ----
        {
            const uint4* src = reinterpret_cast<const uint4*>(
                A + (size_t)(m0 + arow) * K + k0 + acb);
            uint4 a0 = src[0], a1 = src[1];
            *reinterpret_cast<uint4*>(&sA[arow][acb])     = a0;
            *reinterpret_cast<uint4*>(&sA[arow][acb + 8]) = a1;
        }
        // ---- stage B (fp32 -> bf16), stored transposed [n][k] ----
        if (BTRANS == 0) {
            // B tile is 32(k) x 128(n): thread reads 16 contiguous n
            const int k  = tid >> 3;          // 0..31
            const int nb = (tid & 7) * 16;    // 0..112
            const float4* src = reinterpret_cast<const float4*>(
                B + (size_t)(k0 + k) * N + n0 + nb);
            #pragma unroll
            for (int q4 = 0; q4 < 4; q4++) {
                float4 f = src[q4];
                sBt[nb + q4 * 4 + 0][k] = tobf(f.x);
                sBt[nb + q4 * 4 + 1][k] = tobf(f.y);
                sBt[nb + q4 * 4 + 2][k] = tobf(f.z);
                sBt[nb + q4 * 4 + 3][k] = tobf(f.w);
            }
        } else {
            // B tile is 128(n) x 32(k): thread reads 16 contiguous k
            const int n  = tid >> 1;          // 0..127
            const int kb = (tid & 1) * 16;    // 0 or 16
            const float4* src = reinterpret_cast<const float4*>(
                B + (size_t)(n0 + n) * K + k0 + kb);
            float4 f0 = src[0], f1 = src[1], f2 = src[2], f3 = src[3];
            uint4 p0, p1;
            p0.x = pack2bf(f0.x, f0.y); p0.y = pack2bf(f0.z, f0.w);
            p0.z = pack2bf(f1.x, f1.y); p0.w = pack2bf(f1.z, f1.w);
            p1.x = pack2bf(f2.x, f2.y); p1.y = pack2bf(f2.z, f2.w);
            p1.z = pack2bf(f3.x, f3.y); p1.w = pack2bf(f3.z, f3.w);
            *reinterpret_cast<uint4*>(&sBt[n][kb])     = p0;
            *reinterpret_cast<uint4*>(&sBt[n][kb + 8]) = p1;
        }
        // ---- prefetch next tiles (global_prefetch_b8) ----
        if (k0 + BK < K) {
            __builtin_prefetch(A + (size_t)(m0 + arow) * K + (k0 + BK) + acb, 0, 1);
            if (BTRANS == 0)
                __builtin_prefetch(B + (size_t)(k0 + BK + (tid >> 3)) * N + n0 + (tid & 7) * 16, 0, 1);
            else
                __builtin_prefetch(B + (size_t)(n0 + (tid >> 1)) * K + (k0 + BK) + (tid & 1) * 16, 0, 1);
        }
        __syncthreads();

        // ---- 2 A fragments (contiguous k = 16*hi16 + e, as B), rows waveM*32 + {0,16} ----
        v16bf af[2];
        #pragma unroll
        for (int mm = 0; mm < 2; mm++) {
            const int mrow = waveM * 32 + mm * 16 + l16;
            const int kb = hi16 * 16;
            #pragma unroll
            for (int e = 0; e < 8; e++) af[mm][e] = sA[mrow][kb + e];
            #pragma unroll
            for (int e = 0; e < 8; e++) af[mm][8 + e] = sA[mrow][kb + 8 + e];
        }
        // ---- 4 B fragments; 8 WMMAs (each B reused by both A frags) ----
        #pragma unroll
        for (int nn = 0; nn < 4; nn++) {
            v16bf bfv;
            const int ncol = waveN * 64 + nn * 16 + l16;
            const int kb = hi16 * 16;
            #pragma unroll
            for (int e = 0; e < 16; e++) bfv[e] = sBt[ncol][kb + e];
            acc[0][nn] = __builtin_amdgcn_wmma_f32_16x16x32_bf16(
                false, af[0], false, bfv, (short)0, acc[0][nn], false, false);
            acc[1][nn] = __builtin_amdgcn_wmma_f32_16x16x32_bf16(
                false, af[1], false, bfv, (short)0, acc[1][nn], false, false);
        }
        __syncthreads();
    }

    // ---- epilogue (C-layout: lane = col l16, VGPR j = row j + 8*hi16) ----
    #pragma unroll
    for (int mm = 0; mm < 2; mm++) {
        #pragma unroll
        for (int nn = 0; nn < 4; nn++) {
            const int n = n0 + waveN * 64 + nn * 16 + l16;
            #pragma unroll
            for (int j = 0; j < 8; j++) {
                const int m = m0 + waveM * 32 + mm * 16 + j + hi16 * 8;
                const float v = acc[mm][nn][j];
                if (SMODE == 0) {
                    Cf[(size_t)m * N + n] = v;
                } else {
                    const int head = n >> 7, hd = n & 127;
                    Cbh[((size_t)head * M + m) * 128 + hd] = tobf(v);
                }
            }
        }
    }
}

// ---------------------------------------------------------------------------
// Kernel 3: RoPE in-place on per-head bf16 layout [NH][S][HD]
// ---------------------------------------------------------------------------
__global__ __launch_bounds__(256) void rope_inplace_kernel(__bf16* __restrict__ q, int S)
{
    const long long idx = (long long)blockIdx.x * 256 + threadIdx.x;
    const long long total = (long long)NH_ * S * (HD_ / 2);
    if (idx >= total) return;
    const int i = (int)(idx & 63);
    const long long t = idx >> 6;
    const int s = (int)(t % S);
    const int n = (int)(t / S);
    const float inv = __expf(-(2.0f * (float)i / (float)HD_) * 9.210340371976184f);
    const float ang = (float)s * inv;
    const float c = __cosf(ang), sn = __sinf(ang);
    const size_t base = ((size_t)n * S + s) * HD_ + 2 * i;
    const float xr = (float)q[base], xi = (float)q[base + 1];
    q[base]     = tobf(xr * c - xi * sn);
    q[base + 1] = tobf(xr * sn + xi * c);
}

// ---------------------------------------------------------------------------
// Kernel 4: causal flash attention (WMMA bf16, fp32 online softmax).
//   Q/K/V per-head bf16 [NH][S][128].  attn out: bf16 row-major [S][NH*128].
//   Grid (S/64, NH); 128 threads = 4 waves; each wave owns 16 query rows.
//   K tile (32x128) staged by Tensor Data Mover with row padding.
// ---------------------------------------------------------------------------
__global__ __launch_bounds__(128) void flash_attn_kernel(
    const __bf16* __restrict__ qh, const __bf16* __restrict__ kh,
    const __bf16* __restrict__ vh, __bf16* __restrict__ attn, int S)
{
    const int head = blockIdx.y;
    const int qb   = blockIdx.x;                 // 64 query rows per block
    const int tid  = threadIdx.x;
    const int lane = tid & 31;
    const int wave = tid >> 5;
    const int hi16 = lane >> 4;
    const int l16  = lane & 15;

    const __bf16* Q  = qh + (size_t)head * S * HD_;
    const __bf16* Kp = kh + (size_t)head * S * HD_;
    const __bf16* Vp = vh + (size_t)head * S * HD_;

    __shared__ __bf16 sK[32][HD_ + 8];           // [key][hd]; 272B row (TDM pad)
    __shared__ __bf16 sVt[HD_][32 + 8];          // [hd][key] (transposed)
    __shared__ __bf16 sP[4][16][32 + 8];         // per-wave P staging

    const int rowBase = qb * 64 + wave * 16;

    // ---- persistent Q fragments: 4 chunks of 16x32 over head dim ----
    v16bf qf[4];
    {
        const __bf16* qrow = Q + (size_t)(rowBase + l16) * HD_;
        #pragma unroll
        for (int c = 0; c < 4; c++) {
            const int kb = c * 32 + hi16 * 16;
            const v8bf lo = *reinterpret_cast<const v8bf*>(qrow + kb);
            const v8bf hi = *reinterpret_cast<const v8bf*>(qrow + kb + 8);
            #pragma unroll
            for (int e = 0; e < 8; e++) { qf[c][e] = lo[e]; qf[c][8 + e] = hi[e]; }
        }
    }

    v8f o[8] = {};
    float mrow[8], lrow[8];
    #pragma unroll
    for (int j = 0; j < 8; j++) { mrow[j] = -1e30f; lrow[j] = 0.f; }

    const float sm = 0.08838834764831845f;       // 1/sqrt(128)
    const int nkt = qb * 2 + 2;

    for (int kt = 0; kt < nkt; kt++) {
        const int keyBase = kt * 32;
        // ---- stage K tile via TDM (async DMA -> LDS with row pad) ----
#if HAVE_TDM
        if (tid == 0)
            tdm_load_k_tile(Kp + (size_t)keyBase * HD_, lds_byte_off(&sK[0][0]), 32);
#else
        {
            const int key = tid >> 2;
            const int cb  = (tid & 3) * 32;
            const uint4* ks = reinterpret_cast<const uint4*>(
                Kp + (size_t)(keyBase + key) * HD_ + cb);
            uint4 k4[4];
            #pragma unroll
            for (int q4 = 0; q4 < 4; q4++) k4[q4] = ks[q4];
            const __bf16* kb16 = reinterpret_cast<const __bf16*>(k4);
            #pragma unroll
            for (int e = 0; e < 32; e++) sK[key][cb + e] = kb16[e];
        }
#endif
        // ---- stage V tile transposed (vector global load, scalar scatter) ----
        {
            const int key = tid >> 2;
            const int cb  = (tid & 3) * 32;
            const uint4* vs = reinterpret_cast<const uint4*>(
                Vp + (size_t)(keyBase + key) * HD_ + cb);
            uint4 v4[4];
            #pragma unroll
            for (int q4 = 0; q4 < 4; q4++) v4[q4] = vs[q4];
            const __bf16* vb16 = reinterpret_cast<const __bf16*>(v4);
            #pragma unroll
            for (int e = 0; e < 32; e++) sVt[cb + e][key] = vb16[e];
            // prefetch next tile
            if (kt + 1 < nkt) {
                __builtin_prefetch(Kp + (size_t)(keyBase + 32 + key) * HD_ + cb, 0, 1);
                __builtin_prefetch(Vp + (size_t)(keyBase + 32 + key) * HD_ + cb, 0, 1);
            }
        }
#if HAVE_TDM
        if (tid == 0) __builtin_amdgcn_s_wait_tensorcnt(0);
#endif
        __syncthreads();

        // ---- scores: S_tile(16x32) = Q(16x128) * K^T(128x32) ----
        v8f s0 = {}, s1 = {};
        #pragma unroll
        for (int c = 0; c < 4; c++) {
            v16bf b0, b1;
            const int kb = c * 32 + hi16 * 16;
            #pragma unroll
            for (int e = 0; e < 16; e++) b0[e] = sK[l16][kb + e];
            #pragma unroll
            for (int e = 0; e < 16; e++) b1[e] = sK[16 + l16][kb + e];
            s0 = __builtin_amdgcn_wmma_f32_16x16x32_bf16(false, qf[c], false, b0, (short)0, s0, false, false);
            s1 = __builtin_amdgcn_wmma_f32_16x16x32_bf16(false, qf[c], false, b1, (short)0, s1, false, false);
        }

        // ---- scale, causal mask, online softmax ----
        float alpha[8];
        #pragma unroll
        for (int j = 0; j < 8; j++) {
            const int srow = rowBase + j + hi16 * 8;
            float v0 = s0[j] * sm, v1 = s1[j] * sm;
            if (keyBase + l16 > srow)      v0 = -1e30f;
            if (keyBase + 16 + l16 > srow) v1 = -1e30f;
            float t = fmaxf(v0, v1);
            #pragma unroll
            for (int off = 8; off; off >>= 1) t = fmaxf(t, __shfl_xor(t, off, 32));
            const float mnew = fmaxf(mrow[j], t);
            const float a  = __expf(mrow[j] - mnew);
            const float p0 = __expf(v0 - mnew);
            const float p1 = __expf(v1 - mnew);
            float ps = p0 + p1;
            #pragma unroll
            for (int off = 8; off; off >>= 1) ps += __shfl_xor(ps, off, 32);
            lrow[j] = lrow[j] * a + ps;
            mrow[j] = mnew;
            alpha[j] = a;
            s0[j] = p0; s1[j] = p1;
        }
        #pragma unroll
        for (int hc = 0; hc < 8; hc++)
            #pragma unroll
            for (int j = 0; j < 8; j++) o[hc][j] *= alpha[j];

        // ---- P: C-layout -> A-layout via per-wave LDS ----
        #pragma unroll
        for (int j = 0; j < 8; j++) {
            sP[wave][j + hi16 * 8][l16]      = tobf(s0[j]);
            sP[wave][j + hi16 * 8][16 + l16] = tobf(s1[j]);
        }
        v16bf pf;
        {
            const int kb = hi16 * 16;
            #pragma unroll
            for (int e = 0; e < 8; e++) pf[e] = sP[wave][l16][kb + e];
            #pragma unroll
            for (int e = 0; e < 8; e++) pf[8 + e] = sP[wave][l16][kb + 8 + e];
        }
        // ---- O += P(16x32) * V(32x128) ----
        #pragma unroll
        for (int hc = 0; hc < 8; hc++) {
            v16bf bv;
            const int n = hc * 16 + l16;
            const int kb = hi16 * 16;
            #pragma unroll
            for (int e = 0; e < 16; e++) bv[e] = sVt[n][kb + e];
            o[hc] = __builtin_amdgcn_wmma_f32_16x16x32_bf16(false, pf, false, bv, (short)0, o[hc], false, false);
        }
        __syncthreads();
    }

    // ---- normalize and store attn ----
    #pragma unroll
    for (int hc = 0; hc < 8; hc++) {
        const int hd = hc * 16 + l16;
        #pragma unroll
        for (int j = 0; j < 8; j++) {
            const int srow = rowBase + j + hi16 * 8;
            const float v = o[hc][j] / lrow[j];
            attn[(size_t)srow * (NH_ * HD_) + head * HD_ + hd] = tobf(v);
        }
    }
}

// ---------------------------------------------------------------------------
// Launcher
// ---------------------------------------------------------------------------
extern "C" void kernel_launch(void* const* d_in, const int* in_sizes, int n_in,
                              void* d_out, int out_size, void* d_ws, size_t ws_size,
                              hipStream_t stream)
{
    (void)in_sizes; (void)n_in; (void)out_size; (void)ws_size;
    const int S = SEQ_, D = D_;

    const float* xs     = (const float*)d_in[0];
    const float* norm_w = (const float*)d_in[1];
    const float* wq     = (const float*)d_in[2];
    const float* wk     = (const float*)d_in[3];
    const float* wv     = (const float*)d_in[4];
    const float* wo     = (const float*)d_in[5];
    float* out = (float*)d_out;

    // workspace (bf16): h | qh | kh | vh | attn   (5 * S*D*2B = 84 MB)
    __bf16* hbf  = (__bf16*)d_ws;
    __bf16* qhb  = hbf  + (size_t)S * D;
    __bf16* khb  = qhb  + (size_t)S * D;
    __bf16* vhb  = khb  + (size_t)S * D;
    __bf16* attn = vhb  + (size_t)S * D;

    rmsnorm_kernel<<<S, 256, 0, stream>>>(xs, norm_w, hbf, D);

    dim3 ggrid(D / 128, S / 128);
    gemm_bf16_wmma<0, 1><<<ggrid, 256, 0, stream>>>(hbf, wq, nullptr, qhb, S, D, D);
    gemm_bf16_wmma<0, 1><<<ggrid, 256, 0, stream>>>(hbf, wk, nullptr, khb, S, D, D);
    gemm_bf16_wmma<0, 1><<<ggrid, 256, 0, stream>>>(hbf, wv, nullptr, vhb, S, D, D);

    const long long npairs = (long long)NH_ * S * (HD_ / 2);
    const int rblocks = (int)((npairs + 255) / 256);
    rope_inplace_kernel<<<rblocks, 256, 0, stream>>>(qhb, S);
    rope_inplace_kernel<<<rblocks, 256, 0, stream>>>(khb, S);

    flash_attn_kernel<<<dim3(S / 64, NH_), 128, 0, stream>>>(qhb, khb, vhb, attn, S);

    gemm_bf16_wmma<1, 0><<<dim3(D / 128, S / 128), 256, 0, stream>>>(attn, wo, out, nullptr, S, D, D);
}
